// DenoiseStepWrapper_37898791419938
// MI455X (gfx1250) — compile-verified, matrix operand order checked by reading, not run on hardware
//
#include <hip/hip_runtime.h>
#include <math.h>

// ---------------------------------------------------------------------------
// Model constants (from reference)
// ---------------------------------------------------------------------------
#define L_   18
#define H_   1024
#define NH_  8
#define HD_  256
#define FF_  4096
#define AH_  50
#define AD_  32
#define P_   968
#define B_   4
#define MROWS (B_*AH_)      // 200
#define SKEY 1024           // padded key length (P_+AH_=1018 -> 1024)
#define NEGF (-2.3819763e+38f)
#define SCALING (0.0625f)   // HD^-0.5 = 1/16

// WMMA tiling
#define TM 64
#define TN 128
#define TK 32

typedef __attribute__((ext_vector_type(16))) __bf16 v16bf;
typedef __attribute__((ext_vector_type(8)))  __bf16 v8bf;
typedef __attribute__((ext_vector_type(8)))  float  v8f;

union V16U { v16bf v16; v8bf v8[2]; };

// ---------------------------------------------------------------------------
// CDNA5 async global->LDS copy (VGLOBAL op 98, ASYNCcnt-tracked).
// vdst = per-lane LDS byte address, vaddr = per-lane 64-bit global address.
// ---------------------------------------------------------------------------
__device__ inline void async_copy_b128(void* lds_ptr, const void* gptr) {
  unsigned lds_addr = (unsigned)(uintptr_t)lds_ptr;      // low 32 bits = LDS offset
  unsigned long long ga = (unsigned long long)(uintptr_t)gptr;
  asm volatile("global_load_async_to_lds_b128 %0, %1, off"
               :: "v"(lds_addr), "v"(ga)
               : "memory");
}
#define WAIT_ASYNCCNT_0() asm volatile("s_wait_asynccnt 0x0" ::: "memory")
#define WAIT_ASYNCCNT_2() asm volatile("s_wait_asynccnt 0x2" ::: "memory")

// Build a 16-element bf16 WMMA fragment from an LDS row of 32 bf16.
// Per CDNA5 ISA 7.12.2 (16-bit A 16x32): lane m=lane&15, h=lane>>4 holds
// K = h*8 + [0..8) in VGPRs 0..3 and K = 16 + h*8 + [0..8) in VGPRs 4..7.
__device__ inline v16bf make_frag(const __bf16* row32, int h) {
  V16U u;
  u.v8[0] = *(const v8bf*)(row32 + h * 8);
  u.v8[1] = *(const v8bf*)(row32 + 16 + h * 8);
  return u.v16;
}

// Same fragment but sourced from an fp32 LDS row (async-copied raw tile),
// with in-register bf16 conversion.
__device__ inline v16bf make_frag_from_f32(const float* row32, int h) {
  union { v16bf v; __bf16 e[16]; } u;
#pragma unroll
  for (int j = 0; j < 8; ++j) u.e[j] = (__bf16)row32[h * 8 + j];
#pragma unroll
  for (int j = 0; j < 8; ++j) u.e[8 + j] = (__bf16)row32[16 + h * 8 + j];
  return u.v;
}

__device__ inline void wave_mma(const __bf16* lA, const __bf16* lBt,
                                int wm, int wn, int lm, int lh, v8f acc[4]) {
  v16bf af = make_frag(lA + (wm + lm) * TK, lh);
#pragma unroll
  for (int j = 0; j < 4; ++j) {
    v16bf bfj = make_frag(lBt + (wn + j * 16 + lm) * TK, lh);
    acc[j] = __builtin_amdgcn_wmma_f32_16x16x32_bf16(
        false, af, false, bfj, (short)0, acc[j], false, false);
  }
}

__device__ inline float gelu_tanh(float x) {
  float x3 = x * x * x;
  return 0.5f * x * (1.0f + tanhf(0.7978845608028654f * (x + 0.044715f * x3)));
}

// ---------------------------------------------------------------------------
// Generic WMMA GEMM:  C[M,N] (+)= A[M,K] * W[K,N]   (fp32 in, bf16 compute)
//  * A tile staged raw fp32 via GLOBAL_LOAD_ASYNC_TO_LDS_B128, double buffered
//  * W tile prefetched into registers one K-step ahead, cvt->bf16->LDS (transposed)
// EPI 0: C = acc
// EPI 1: C = acc + e0[n]                       (bias)
// EPI 2: C = C + e0[(m/ah)*N + n] * acc        (gated residual, in place)
// EPI 3: C = gelu(e0[m*ldc+n]) * acc           (gelu-mult; e0 may alias C)
// Requires K % 32 == 0.
// ---------------------------------------------------------------------------
template <int EPI>
__global__ __launch_bounds__(256)
void gemm_bf16_wmma(const float* __restrict__ A, const float* __restrict__ W,
                    float* __restrict__ C, int M, int N, int K,
                    int lda, int ldb, int ldc,
                    const float* __restrict__ e0, int ahdiv) {
  __shared__ __align__(16) float  lAf[2][TM * TK];   // fp32 A tiles (async dest)
  __shared__ __align__(16) __bf16 lBt[2][TN * TK];   // bf16 W^T tiles
  const int tid = threadIdx.x;
  const int m0 = blockIdx.y * TM;
  const int n0 = blockIdx.x * TN;
  const int wave = tid >> 5, lane = tid & 31;
  const int wm = (wave & 3) * 16;   // 4 M strips of 16
  const int wn = (wave >> 2) * 64;  // 2 N strips of 64
  const int lm = lane & 15, lh = lane >> 4;

  const int b_k = tid >> 3;          // 0..31  (K row within tile)
  const int b_n = (tid & 7) * 16;    // 0..112 (N chunk within tile)

  // A async assignment: 512 16B chunks / 256 threads = 2 per thread
  auto issueA = [&](int k0, int buf) {
#pragma unroll
    for (int j = 0; j < 2; ++j) {
      int c = tid * 2 + j;
      int row = c >> 3, col4 = (c & 7) * 4;
      int gm = m0 + row;
      float* ldsp = &lAf[buf][row * TK + col4];
      if (gm < M) {
        async_copy_b128(ldsp, A + (size_t)gm * lda + k0 + col4);
      } else {
        ldsp[0] = 0.0f; ldsp[1] = 0.0f; ldsp[2] = 0.0f; ldsp[3] = 0.0f;
      }
    }
  };
  auto loadB = [&](int k0, float* vr) {
    const float* src = W + (size_t)(k0 + b_k) * ldb + n0 + b_n;
#pragma unroll
    for (int j = 0; j < 16; ++j) vr[j] = (n0 + b_n + j < N) ? src[j] : 0.0f;
  };

  float vr[16];
  issueA(0, 0);
  loadB(0, vr);
  v8f acc[4] = {};
  int buf = 0;

  for (int k0 = 0; k0 < K; k0 += TK) {
    // stage current W tile (regs -> bf16 LDS, transposed)
#pragma unroll
    for (int j = 0; j < 16; ++j) lBt[buf][(b_n + j) * TK + b_k] = (__bf16)vr[j];

    const bool more = (k0 + TK) < K;
    if (more) {
      issueA(k0 + TK, buf ^ 1);     // next A tile: async, lands during compute
      WAIT_ASYNCCNT_2();            // current A tile complete (in-order)
    } else {
      WAIT_ASYNCCNT_0();
    }
    __syncthreads();
    if (more) loadB(k0 + TK, vr);   // next W tile: overlap HBM with WMMA

    // compute from buffer `buf`
    v16bf af = make_frag_from_f32(&lAf[buf][(wm + lm) * TK], lh);
#pragma unroll
    for (int j = 0; j < 4; ++j) {
      v16bf bfj = make_frag(&lBt[buf][(wn + j * 16 + lm) * TK], lh);
      acc[j] = __builtin_amdgcn_wmma_f32_16x16x32_bf16(
          false, af, false, bfj, (short)0, acc[j], false, false);
    }
    __syncthreads();
    buf ^= 1;
  }

  // C/D layout: VGPR r, lanes 0-15 -> M=r, lanes 16-31 -> M=8+r; N = lane&15.
#pragma unroll
  for (int r = 0; r < 8; ++r) {
    int gm = m0 + wm + r + lh * 8;
    if (gm >= M) continue;
#pragma unroll
    for (int j = 0; j < 4; ++j) {
      int gn = n0 + wn + j * 16 + lm;
      if (gn >= N) continue;
      float a = acc[j][r];
      size_t idx = (size_t)gm * ldc + gn;
      if (EPI == 0) C[idx] = a;
      else if (EPI == 1) C[idx] = a + e0[gn];
      else if (EPI == 2) C[idx] = C[idx] + e0[(size_t)(gm / ahdiv) * N + gn] * a;
      else if (EPI == 3) C[idx] = gelu_tanh(e0[idx]) * a;
    }
  }
}

// ---------------------------------------------------------------------------
// Attention scores:  S[b,h,m,kk] = (q[b,h,m,:] . k[kk,:]) * SCALING + bias
// grid: (SKEY/TN=8, 1, B*NH)
// ---------------------------------------------------------------------------
__global__ __launch_bounds__(256)
void attn_scores(const float* __restrict__ q, const float* __restrict__ pk,
                 const float* __restrict__ kbuf,
                 const unsigned char* __restrict__ pad,
                 float* __restrict__ S, int layer) {
  __shared__ __align__(16) __bf16 lA[TM * TK];
  __shared__ __align__(16) __bf16 lBt[TN * TK];
  const int tid = threadIdx.x;
  const int bh = blockIdx.z, b = bh / NH_, h = bh % NH_;
  const int n0 = blockIdx.x * TN;
  const int wave = tid >> 5, lane = tid & 31;
  const int wm = (wave & 3) * 16, wn = (wave >> 2) * 64;
  const int lm = lane & 15, lh = lane >> 4;

  const float* Abase = q + (size_t)b * AH_ * (NH_ * HD_) + h * HD_;
  const int a_row = tid >> 2, a_col = (tid & 3) * 8;
  const int b_key = tid >> 1;            // 0..127 key within tile
  const int b_d   = (tid & 1) * 16;      // 0 or 16

  v8f acc[4] = {};

  for (int k0 = 0; k0 < HD_; k0 += TK) {
    {  // A = q tile (rows >= AH_ zeroed)
      __bf16* dst = &lA[a_row * TK + a_col];
      if (a_row < AH_) {
        const float* src = Abase + (size_t)a_row * (NH_ * HD_) + k0 + a_col;
#pragma unroll
        for (int j = 0; j < 8; ++j) dst[j] = (__bf16)src[j];
      } else {
#pragma unroll
        for (int j = 0; j < 8; ++j) dst[j] = (__bf16)0.0f;
      }
    }
    {  // B = K rows: lBt[key][d]
      int kk = n0 + b_key;
      const float* src;
      bool valid = true;
      if (kk < P_)
        src = pk + (((size_t)b * L_ + layer) * P_ + kk) * HD_ + k0 + b_d;
      else if (kk < P_ + AH_)
        src = kbuf + ((size_t)b * AH_ + (kk - P_)) * HD_ + k0 + b_d;
      else { src = nullptr; valid = false; }
#pragma unroll
      for (int j = 0; j < 16; ++j)
        lBt[b_key * TK + b_d + j] = (__bf16)(valid ? src[j] : 0.0f);
    }
    __syncthreads();
    wave_mma(lA, lBt, wm, wn, lm, lh, acc);
    __syncthreads();
  }

#pragma unroll
  for (int r = 0; r < 8; ++r) {
    int gm = wm + r + lh * 8;
    if (gm >= AH_) continue;
#pragma unroll
    for (int j = 0; j < 4; ++j) {
      int kk = n0 + wn + j * 16 + lm;
      float bias;
      if (kk < P_)            bias = pad[(size_t)b * P_ + kk] ? 0.0f : NEGF;
      else if (kk < P_ + AH_) bias = 0.0f;   // suffix mask is all-visible
      else                    bias = NEGF;   // padding cols
      S[((size_t)bh * AH_ + gm) * SKEY + kk] = acc[j][r] * SCALING + bias;
    }
  }
}

// ---------------------------------------------------------------------------
// attn = softmax(S) @ V.  grid: (HD_/TN=2, 1, B*NH). K = SKEY (padded).
// ---------------------------------------------------------------------------
__global__ __launch_bounds__(256)
void attn_v(const float* __restrict__ S, const float* __restrict__ pv,
            const float* __restrict__ vbuf, float* __restrict__ attnb,
            int layer) {
  __shared__ __align__(16) __bf16 lA[TM * TK];
  __shared__ __align__(16) __bf16 lBt[TN * TK];
  const int tid = threadIdx.x;
  const int bh = blockIdx.z, b = bh / NH_, h = bh % NH_;
  const int n0 = blockIdx.x * TN;   // d offset within head
  const int wave = tid >> 5, lane = tid & 31;
  const int wm = (wave & 3) * 16, wn = (wave >> 2) * 64;
  const int lm = lane & 15, lh = lane >> 4;

  const float* Abase = S + (size_t)bh * AH_ * SKEY;
  const int a_row = tid >> 2, a_col = (tid & 3) * 8;
  const int b_k = tid >> 3;          // key within K-tile, 0..31
  const int b_n = (tid & 7) * 16;    // d within N-tile

  v8f acc[4] = {};

  for (int k0 = 0; k0 < SKEY; k0 += TK) {
    {
      __bf16* dst = &lA[a_row * TK + a_col];
      if (a_row < AH_) {
        const float* src = Abase + (size_t)a_row * SKEY + k0 + a_col;
#pragma unroll
        for (int j = 0; j < 8; ++j) dst[j] = (__bf16)src[j];
      } else {
#pragma unroll
        for (int j = 0; j < 8; ++j) dst[j] = (__bf16)0.0f;
      }
    }
    {  // B = V[key][d], transposed into lBt[d][key]
      int kk = k0 + b_k;
      const float* src;
      bool valid = true;
      if (kk < P_)
        src = pv + (((size_t)b * L_ + layer) * P_ + kk) * HD_ + n0 + b_n;
      else if (kk < P_ + AH_)
        src = vbuf + ((size_t)b * AH_ + (kk - P_)) * HD_ + n0 + b_n;
      else { src = nullptr; valid = false; }
#pragma unroll
      for (int j = 0; j < 16; ++j)
        lBt[(b_n + j) * TK + b_k] = (__bf16)(valid ? src[j] : 0.0f);
    }
    __syncthreads();
    wave_mma(lA, lBt, wm, wn, lm, lh, acc);
    __syncthreads();
  }

#pragma unroll
  for (int r = 0; r < 8; ++r) {
    int gm = wm + r + lh * 8;
    if (gm >= AH_) continue;
#pragma unroll
    for (int j = 0; j < 4; ++j) {
      int gn = n0 + wn + j * 16 + lm;
      if (gn >= HD_) continue;
      attnb[((size_t)b * AH_ + gm) * (NH_ * HD_) + h * HD_ + gn] = acc[j][r];
    }
  }
}

// ---------------------------------------------------------------------------
// Helpers
// ---------------------------------------------------------------------------
__device__ inline float wave_max32(float v) {
#pragma unroll
  for (int i = 16; i > 0; i >>= 1) v = fmaxf(v, __shfl_xor(v, i, 32));
  return v;
}
__device__ inline float wave_sum32(float v) {
#pragma unroll
  for (int i = 16; i > 0; i >>= 1) v += __shfl_xor(v, i, 32);
  return v;
}

// Row softmax over SKEY=1024 (grid = B*NH*AH rows, 256 threads)
__global__ __launch_bounds__(256)
void softmax_rows(float* __restrict__ S) {
  float* p = S + (size_t)blockIdx.x * SKEY;
  const int tid = threadIdx.x;
  __shared__ float red[8];
  float v[4], vmax = NEGF;
#pragma unroll
  for (int j = 0; j < 4; ++j) { v[j] = p[tid + j * 256]; vmax = fmaxf(vmax, v[j]); }
  vmax = wave_max32(vmax);
  if ((tid & 31) == 0) red[tid >> 5] = vmax;
  __syncthreads();
  vmax = red[0];
#pragma unroll
  for (int w = 1; w < 8; ++w) vmax = fmaxf(vmax, red[w]);
  __syncthreads();
  float s = 0.0f;
#pragma unroll
  for (int j = 0; j < 4; ++j) { v[j] = expf(v[j] - vmax); s += v[j]; }
  s = wave_sum32(s);
  if ((tid & 31) == 0) red[tid >> 5] = s;
  __syncthreads();
  s = 0.0f;
#pragma unroll
  for (int w = 0; w < 8; ++w) s += red[w];
  float inv = 1.0f / s;
#pragma unroll
  for (int j = 0; j < 4; ++j) p[tid + j * 256] = v[j] * inv;
}

// AdaRMS: y = rmsnorm(x_row) * (1 + scale[b, :])  (rows = B*AH, H=1024)
__global__ __launch_bounds__(256)
void ada_rms(const float* __restrict__ x, const float* __restrict__ scale,
             float* __restrict__ y) {
  const int row = blockIdx.x, b = row / AH_, tid = threadIdx.x;
  const float* xr = x + (size_t)row * H_;
  __shared__ float red[8];
  float ss = 0.0f, v[4];
#pragma unroll
  for (int j = 0; j < 4; ++j) { v[j] = xr[tid + j * 256]; ss += v[j] * v[j]; }
  ss = wave_sum32(ss);
  if ((tid & 31) == 0) red[tid >> 5] = ss;
  __syncthreads();
  ss = 0.0f;
#pragma unroll
  for (int w = 0; w < 8; ++w) ss += red[w];
  float rs = rsqrtf(ss * (1.0f / H_) + 1e-6f);
#pragma unroll
  for (int j = 0; j < 4; ++j) {
    int n = tid + j * 256;
    y[(size_t)row * H_ + n] = v[j] * rs * (1.0f + scale[(size_t)b * H_ + n]);
  }
}

// RoPE in place on (B*AH, nheads, 256) strided buffer
__global__ void rope_apply(float* __restrict__ buf, const int* __restrict__ offs,
                           int nheads, int ld) {
  int idx = blockIdx.x * blockDim.x + threadIdx.x;
  int total = MROWS * nheads * 128;
  if (idx >= total) return;
  int i = idx & 127;
  int t = idx >> 7;
  int hh = t % nheads;
  int row = t / nheads;
  int b = row / AH_, m = row % AH_;
  float pos = (float)(offs[b] + m);
  float inv = powf(10000.0f, -(float)(2 * i) / (float)HD_);
  float f = pos * inv, c = cosf(f), s = sinf(f);
  float* p = buf + (size_t)row * ld + hh * HD_;
  float x1 = p[i], x2 = p[i + 128];
  p[i]       = x1 * c - x2 * s;
  p[i + 128] = x2 * c + x1 * s;
}

// offsets[b] = popcount(pad_mask[b,:])
__global__ __launch_bounds__(256)
void pad_count(const unsigned char* __restrict__ pad, int* __restrict__ offs) {
  const int b = blockIdx.x, tid = threadIdx.x;
  __shared__ int sh[256];
  int s = 0;
  for (int i = tid; i < P_; i += 256) s += pad[(size_t)b * P_ + i] ? 1 : 0;
  sh[tid] = s;
  __syncthreads();
  for (int st = 128; st > 0; st >>= 1) { if (tid < st) sh[tid] += sh[tid + st]; __syncthreads(); }
  if (tid == 0) offs[b] = sh[0];
}

// sinusoidal time features: emb (B, H)
__global__ void time_feats(const float* __restrict__ t, float* __restrict__ emb) {
  int idx = blockIdx.x * blockDim.x + threadIdx.x;
  if (idx >= B_ * H_) return;
  int b = idx / H_, j = idx % H_;
  int i = (j < 512) ? j : j - 512;
  float frac = (float)i / 511.0f;
  float period = 0.004f * powf(1000.0f, frac);
  float sf = 6.283185307179586f / period;
  float v = sf * t[b];
  emb[idx] = (j < 512) ? sinf(v) : cosf(v);
}

// tiny GEMM (M=4): Y = X(4,K) @ W(K,N); epi 0 none, 1 +bias, 2 silu(.+bias)
__global__ __launch_bounds__(256)
void gemm_small(const float* __restrict__ X, const float* __restrict__ W,
                const float* __restrict__ bias, float* __restrict__ Y,
                int K, int N, int epi) {
  int idx = blockIdx.x * blockDim.x + threadIdx.x;
  if (idx >= B_ * N) return;
  int r = idx / N, n = idx % N;
  float acc = 0.0f;
  for (int k = 0; k < K; ++k) acc += X[(size_t)r * K + k] * W[(size_t)k * N + n];
  if (epi >= 1) acc += bias[n];
  if (epi == 2) acc = acc / (1.0f + expf(-acc));
  Y[idx] = acc;
}

// ---------------------------------------------------------------------------
// Launcher.  Input order = setup_inputs() dict order, params dict flattened:
// 0 prefix_keys 1 prefix_values 2 prefix_pad_masks 3 x_t 4 timestep,
// 5 Wq 6 Wk 7 Wv 8 Wo 9 Wg 10 Wu 11 Wd 12 ln1_s 13 ln1_g 14 ln2_s 15 ln2_g
// 16 lnf_s 17 act_in_w 18 act_in_b 19 t_in_w 20 t_in_b 21 t_out_w 22 t_out_b
// 23 act_out_w 24 act_out_b
// ---------------------------------------------------------------------------
extern "C" void kernel_launch(void* const* d_in, const int* in_sizes, int n_in,
                              void* d_out, int out_size, void* d_ws, size_t ws_size,
                              hipStream_t stream) {
  const float* pk  = (const float*)d_in[0];
  const float* pv  = (const float*)d_in[1];
  const unsigned char* pad = (const unsigned char*)d_in[2];
  const float* x_t = (const float*)d_in[3];
  const float* ts  = (const float*)d_in[4];
  const float* Wq  = (const float*)d_in[5];
  const float* Wk  = (const float*)d_in[6];
  const float* Wv  = (const float*)d_in[7];
  const float* Wo  = (const float*)d_in[8];
  const float* Wg  = (const float*)d_in[9];
  const float* Wu  = (const float*)d_in[10];
  const float* Wd  = (const float*)d_in[11];
  const float* ln1s = (const float*)d_in[12];
  const float* ln1g = (const float*)d_in[13];
  const float* ln2s = (const float*)d_in[14];
  const float* ln2g = (const float*)d_in[15];
  const float* lnfs = (const float*)d_in[16];
  const float* aiw = (const float*)d_in[17];
  const float* aib = (const float*)d_in[18];
  const float* tiw = (const float*)d_in[19];
  const float* tib = (const float*)d_in[20];
  const float* tow = (const float*)d_in[21];
  const float* tob = (const float*)d_in[22];
  const float* aow = (const float*)d_in[23];
  const float* aob = (const float*)d_in[24];
  float* out = (float*)d_out;

  // workspace carve-up
  char* cur = (char*)d_ws;
  auto alloc = [&](size_t bytes) {
    char* p = cur;
    cur += (bytes + 255) & ~(size_t)255;
    return p;
  };
  float* emb1   = (float*)alloc(B_ * H_ * 4);
  float* emb2   = (float*)alloc(B_ * H_ * 4);
  float* cond   = (float*)alloc(B_ * H_ * 4);
  float* sc1    = (float*)alloc(B_ * H_ * 4);
  float* g1     = (float*)alloc(B_ * H_ * 4);
  float* sc2    = (float*)alloc(B_ * H_ * 4);
  float* g2     = (float*)alloc(B_ * H_ * 4);
  float* scF    = (float*)alloc(B_ * H_ * 4);
  int*   offs   = (int*)  alloc(B_ * 4);
  float* hidden = (float*)alloc((size_t)MROWS * H_ * 4);
  float* normed = (float*)alloc((size_t)MROWS * H_ * 4);
  float* qbuf   = (float*)alloc((size_t)MROWS * NH_ * HD_ * 4);
  float* kbuf   = (float*)alloc((size_t)MROWS * HD_ * 4);
  float* vbuf   = (float*)alloc((size_t)MROWS * HD_ * 4);
  float* Sbuf   = (float*)alloc((size_t)B_ * NH_ * AH_ * SKEY * 4);
  float* attnb  = (float*)alloc((size_t)MROWS * NH_ * HD_ * 4);
  float* Gbuf   = (float*)alloc((size_t)MROWS * FF_ * 4);
  (void)in_sizes; (void)n_in; (void)out_size; (void)ws_size;

  const dim3 blk(256);
  const int MG = (MROWS + TM - 1) / TM;  // 4

  // --- conditioning ---
  time_feats<<<(B_ * H_ + 255) / 256, blk, 0, stream>>>(ts, emb1);
  gemm_small<<<(B_ * H_ + 255) / 256, blk, 0, stream>>>(emb1, tiw, tib, emb2, H_, H_, 2);
  gemm_small<<<(B_ * H_ + 255) / 256, blk, 0, stream>>>(emb2, tow, tob, cond, H_, H_, 2);
  pad_count<<<B_, blk, 0, stream>>>(pad, offs);

  // hidden = x_t @ act_in_w + b     (200x32 @ 32x1024)
  gemm_bf16_wmma<1><<<dim3(H_ / TN, MG), blk, 0, stream>>>(
      x_t, aiw, hidden, MROWS, H_, AD_, AD_, H_, H_, aib, 0);

  for (int l = 0; l < L_; ++l) {
    const size_t oHH = (size_t)l * H_ * H_;
    gemm_small<<<(B_ * H_ + 255) / 256, blk, 0, stream>>>(cond, ln1s + oHH, nullptr, sc1, H_, H_, 0);
    gemm_small<<<(B_ * H_ + 255) / 256, blk, 0, stream>>>(cond, ln1g + oHH, nullptr, g1, H_, H_, 0);
    ada_rms<<<MROWS, blk, 0, stream>>>(hidden, sc1, normed);

    // q/k/v projections
    gemm_bf16_wmma<0><<<dim3(NH_ * HD_ / TN, MG), blk, 0, stream>>>(
        normed, Wq + (size_t)l * H_ * NH_ * HD_, qbuf,
        MROWS, NH_ * HD_, H_, H_, NH_ * HD_, NH_ * HD_, nullptr, 0);
    gemm_bf16_wmma<0><<<dim3(HD_ / TN, MG), blk, 0, stream>>>(
        normed, Wk + (size_t)l * H_ * HD_, kbuf,
        MROWS, HD_, H_, H_, HD_, HD_, nullptr, 0);
    gemm_bf16_wmma<0><<<dim3(HD_ / TN, MG), blk, 0, stream>>>(
        normed, Wv + (size_t)l * H_ * HD_, vbuf,
        MROWS, HD_, H_, H_, HD_, HD_, nullptr, 0);

    rope_apply<<<(MROWS * NH_ * 128 + 255) / 256, blk, 0, stream>>>(qbuf, offs, NH_, NH_ * HD_);
    rope_apply<<<(MROWS * 1 * 128 + 255) / 256, blk, 0, stream>>>(kbuf, offs, 1, HD_);

    // attention
    attn_scores<<<dim3(SKEY / TN, 1, B_ * NH_), blk, 0, stream>>>(qbuf, pk, kbuf, pad, Sbuf, l);
    softmax_rows<<<B_ * NH_ * AH_, blk, 0, stream>>>(Sbuf);
    attn_v<<<dim3(HD_ / TN, 1, B_ * NH_), blk, 0, stream>>>(Sbuf, pv, vbuf, attnb, l);

    // hidden += gate1 * (attn @ Wo)
    gemm_bf16_wmma<2><<<dim3(H_ / TN, MG), blk, 0, stream>>>(
        attnb, Wo + (size_t)l * NH_ * HD_ * H_, hidden,
        MROWS, H_, NH_ * HD_, NH_ * HD_, H_, H_, g1, AH_);

    gemm_small<<<(B_ * H_ + 255) / 256, blk, 0, stream>>>(cond, ln2s + oHH, nullptr, sc2, H_, H_, 0);
    gemm_small<<<(B_ * H_ + 255) / 256, blk, 0, stream>>>(cond, ln2g + oHH, nullptr, g2, H_, H_, 0);
    ada_rms<<<MROWS, blk, 0, stream>>>(hidden, sc2, normed);

    // G = normed @ Wg ; G = gelu(G) * (normed @ Wu) ; hidden += gate2 * (G @ Wd)
    gemm_bf16_wmma<0><<<dim3(FF_ / TN, MG), blk, 0, stream>>>(
        normed, Wg + (size_t)l * H_ * FF_, Gbuf,
        MROWS, FF_, H_, H_, FF_, FF_, nullptr, 0);
    gemm_bf16_wmma<3><<<dim3(FF_ / TN, MG), blk, 0, stream>>>(
        normed, Wu + (size_t)l * H_ * FF_, Gbuf,
        MROWS, FF_, H_, H_, FF_, FF_, Gbuf, 0);
    gemm_bf16_wmma<2><<<dim3(H_ / TN, MG), blk, 0, stream>>>(
        Gbuf, Wd + (size_t)l * FF_ * H_, hidden,
        MROWS, H_, FF_, FF_, H_, H_, g2, AH_);
  }

  // final norm + act_out
  gemm_small<<<(B_ * H_ + 255) / 256, blk, 0, stream>>>(cond, lnfs, nullptr, scF, H_, H_, 0);
  ada_rms<<<MROWS, blk, 0, stream>>>(hidden, scF, normed);
  gemm_bf16_wmma<1><<<dim3(1, MG), blk, 0, stream>>>(
      normed, aow, out, MROWS, AD_, H_, H_, AD_, AD_, aob, 0);
}
